// GATConv_3719441678945
// MI455X (gfx1250) — matrix operand order repaired and checked by reading, not compiled, not run on hardware
//
#include <hip/hip_runtime.h>
#include <hip/hip_bf16.h>

typedef __attribute__((ext_vector_type(16))) _Float16 v16h;
typedef __attribute__((ext_vector_type(8)))  _Float16 v8h;
typedef __attribute__((ext_vector_type(8)))  float    v8f;

#define NEG_SLOPE 0.2f

// Native CDNA5 f32 atomic add, non-returning (tracked by STOREcnt), device
// scope so the RMW executes at L2 (where the 51MB accumulator is resident).
// Guarantees global_atomic_add_f32 instead of a possible CAS-loop lowering.
__device__ __forceinline__ void atomic_add_f32_dev(float* p, float v) {
  asm volatile("global_atomic_add_f32 %0, %1, off scope:SCOPE_DEV"
               :: "v"(p), "v"(v)
               : "memory");
}

// ---------------------------------------------------------------------------
// K0a: reduced attention weights.
// w_src_red[k][h] = sum_d W_fc[k][h*32+d] * attn_src[h][d]   (k<128)
// w_dst_red likewise; w_edge_red[k][h] from W_edge/attn_edge (k<16)
// ---------------------------------------------------------------------------
__global__ __launch_bounds__(256) void gat_reduce_attn_w(
    const float* __restrict__ W_fc, const float* __restrict__ asrc,
    const float* __restrict__ adst, const float* __restrict__ W_edge,
    const float* __restrict__ aedge, float* __restrict__ wsrc,
    float* __restrict__ wdst, float* __restrict__ wedge)
{
  int t = blockIdx.x * 256 + threadIdx.x;
  if (t < 512) {
    int k = t >> 2, h = t & 3;
    float ss = 0.f, dd = 0.f;
    for (int d = 0; d < 32; ++d) {
      float w = W_fc[k * 128 + h * 32 + d];
      ss += w * asrc[h * 32 + d];
      dd += w * adst[h * 32 + d];
    }
    wsrc[t] = ss; wdst[t] = dd;
  } else if (t < 512 + 64) {
    int u = t - 512;
    int k = u >> 2, h = u & 3;
    float s2 = 0.f;
    for (int d = 0; d < 32; ++d)
      s2 += W_edge[k * 128 + h * 32 + d] * aedge[h * 32 + d];
    wedge[u] = s2;
  }
}

// ---------------------------------------------------------------------------
// K0b: node_inputs f32 -> f16 (row-major, feeds WMMA A operand directly)
// ---------------------------------------------------------------------------
__global__ __launch_bounds__(256) void gat_cvt_f16(
    const float* __restrict__ x, _Float16* __restrict__ xh, int n)
{
  int i = blockIdx.x * 256 + threadIdx.x;
  if (i < n) xh[i] = (_Float16)x[i];
}

// ---------------------------------------------------------------------------
// K0c: pack W_fc / W_res (128x128 f32, row-major [K][N]) into per-lane WMMA
// B-layout f16: P[((tn*4+ks)*32 + lane)*16 + j] = W[ks*32 + (lane>>4)*8 + (j&7) + (j>>3)*16][tn*16 + (lane&15)]
// so each lane's B fragment is one contiguous 32-byte chunk.
// ---------------------------------------------------------------------------
__global__ __launch_bounds__(256) void gat_pack_w(
    const float* __restrict__ Wfc, const float* __restrict__ Wres,
    _Float16* __restrict__ Pfc, _Float16* __restrict__ Pres)
{
  int t = blockIdx.x * 256 + threadIdx.x;
  if (t >= 32768) return;
  const float* W = (t < 16384) ? Wfc : Wres;
  _Float16*    P = (t < 16384) ? Pfc : Pres;
  int u    = t & 16383;
  int j    = u & 15;
  int lane = (u >> 4) & 31;
  int ks   = (u >> 9) & 3;
  int tn   = u >> 11;
  int n  = tn * 16 + (lane & 15);
  int K  = ks * 32 + ((lane >> 4) * 8) + (j & 7) + ((j >> 3) * 16);
  P[u] = (_Float16)W[K * 128 + n];
}

// ---------------------------------------------------------------------------
// K1: WMMA node projection. One block = 8 waves = one 16-row M tile.
// Waves 0-3: W_fc -> feat (two 16-col N tiles each).
// Waves 4-7: W_res -> d_out (+bias).
// ---------------------------------------------------------------------------
__global__ __launch_bounds__(256) void gat_node_proj_wmma(
    const _Float16* __restrict__ Ah,   // [N][128] f16
    const _Float16* __restrict__ Pfc,  // packed B
    const _Float16* __restrict__ Pres, // packed B
    const float* __restrict__ bias,    // [128]
    float* __restrict__ feat,          // [N][128]
    float* __restrict__ outx,          // [N][128] (d_out)
    int N)
{
  const int lane  = threadIdx.x & 31;
  const int wave  = threadIdx.x >> 5;
  const int mbase = blockIdx.x * 16;
  if (mbase >= N) return;

  const bool is_res = (wave >= 4);
  const int  nt0    = (wave & 3) * 2;                 // first of two N tiles
  const _Float16* __restrict__ W = is_res ? Pres : Pfc;

  const int row = lane & 15;
  const int kb  = (lane >> 4) * 8;                    // A lane K-base: 0 or 8
  int arowi = mbase + row;
  if (arowi >= N) arowi = N - 1;                      // clamp (stores guarded)
  const _Float16* __restrict__ arow = Ah + (size_t)arowi * 128;

  v8f acc0 = {}; v8f acc1 = {};

#pragma unroll
  for (int ks = 0; ks < 4; ++ks) {
    const int k0 = ks * 32;
    // A fragment: K = kb..kb+7 then kb+16..kb+23 (two contiguous 8-half runs)
    v8h lo = *(const v8h*)(arow + k0 + kb);
    v8h hi = *(const v8h*)(arow + k0 + kb + 16);
    v16h a;
#pragma unroll
    for (int i = 0; i < 8; ++i) { a[i] = lo[i]; a[8 + i] = hi[i]; }

    const v16h b0 = *(const v16h*)(W + ((size_t)((nt0 + 0) * 4 + ks) * 32 + lane) * 16);
    const v16h b1 = *(const v16h*)(W + ((size_t)((nt0 + 1) * 4 + ks) * 32 + lane) * 16);

    acc0 = __builtin_amdgcn_wmma_f32_16x16x32_f16(false, a, false, b0,
                                                  (short)0, acc0, false, false);
    acc1 = __builtin_amdgcn_wmma_f32_16x16x32_f16(false, a, false, b1,
                                                  (short)0, acc1, false, false);
  }

  // C layout: VGPR i: lanes 0-15 -> (M=i, N=lane), lanes 16-31 -> (M=8+i, N=lane-16)
  float* __restrict__ dstp = is_res ? outx : feat;
  const int col0 = nt0 * 16 + (lane & 15);
  const int r    = mbase + (lane >> 4) * 8;
  const float b0add = is_res ? bias[col0]      : 0.f;
  const float b1add = is_res ? bias[col0 + 16] : 0.f;
#pragma unroll
  for (int i = 0; i < 8; ++i) {
    if (r + i < N) {
      dstp[(size_t)(r + i) * 128 + col0]      = acc0[i] + b0add;
      dstp[(size_t)(r + i) * 128 + col0 + 16] = acc1[i] + b1add;
    }
  }
}

// ---------------------------------------------------------------------------
// K2: per-node logits a_src/a_dst via reduced weights; one wave per node.
// Also initializes emax (ordered-uint encoding, 0 == -inf) and denom.
// ---------------------------------------------------------------------------
__global__ __launch_bounds__(256) void gat_node_logits(
    const float* __restrict__ x,     // node_inputs [N][128]
    const float* __restrict__ wsrc,  // [128][4]
    const float* __restrict__ wdst,  // [128][4]
    float* __restrict__ a_src, float* __restrict__ a_dst,
    unsigned* __restrict__ emax, float* __restrict__ denom, int N)
{
  int node = blockIdx.x * 8 + (threadIdx.x >> 5);
  int lane = threadIdx.x & 31;
  if (node >= N) return;

  const float4 xv = *(const float4*)(x + (size_t)node * 128 + lane * 4);
  float xs[4] = {xv.x, xv.y, xv.z, xv.w};
  float s[4] = {0.f, 0.f, 0.f, 0.f};
  float d[4] = {0.f, 0.f, 0.f, 0.f};
#pragma unroll
  for (int j = 0; j < 4; ++j) {
    int k = lane * 4 + j;
    float xk = xs[j];
#pragma unroll
    for (int h = 0; h < 4; ++h) {
      s[h] += xk * wsrc[k * 4 + h];
      d[h] += xk * wdst[k * 4 + h];
    }
  }
#pragma unroll
  for (int off = 16; off > 0; off >>= 1) {
#pragma unroll
    for (int h = 0; h < 4; ++h) {
      s[h] += __shfl_xor(s[h], off, 32);
      d[h] += __shfl_xor(d[h], off, 32);
    }
  }
  if (lane == 0) {
#pragma unroll
    for (int h = 0; h < 4; ++h) {
      a_src[node * 4 + h] = s[h];
      a_dst[node * 4 + h] = d[h];
      emax[node * 4 + h]  = 0u;    // ordered-uint -inf
      denom[node * 4 + h] = 0.f;
    }
  }
}

__device__ __forceinline__ unsigned f32_to_ordered(float v) {
  unsigned b = __float_as_uint(v);
  return b ^ ((unsigned)((int)b >> 31) | 0x80000000u);
}
__device__ __forceinline__ float ordered_to_f32(unsigned key) {
  unsigned mask = (unsigned)((int)(~key) >> 31) | 0x80000000u;
  return __uint_as_float(key ^ mask);
}

// ---------------------------------------------------------------------------
// K3: per-edge scores + LeakyReLU, atomic segment-max by dst.
// ---------------------------------------------------------------------------
__global__ __launch_bounds__(256) void gat_edge_logits(
    const float* __restrict__ ein, const int* __restrict__ src,
    const int* __restrict__ dst, const float* __restrict__ wedge, // [16][4]
    const float* __restrict__ a_src, const float* __restrict__ a_dst,
    float* __restrict__ escore, unsigned* __restrict__ emax, int E)
{
  int e = blockIdx.x * 256 + threadIdx.x;
  if (e >= E) return;
  const float* __restrict__ x = ein + (size_t)e * 16;
  float ae[4] = {0.f, 0.f, 0.f, 0.f};
#pragma unroll
  for (int k = 0; k < 16; ++k) {
    float xk = x[k];
#pragma unroll
    for (int h = 0; h < 4; ++h) ae[h] += xk * wedge[k * 4 + h];
  }
  int s = src[e], d = dst[e];
#pragma unroll
  for (int h = 0; h < 4; ++h) {
    float v = a_src[s * 4 + h] + a_dst[d * 4 + h] + ae[h];
    v = (v > 0.f) ? v : NEG_SLOPE * v;
    escore[(size_t)e * 4 + h] = v;
    atomicMax(&emax[d * 4 + h], f32_to_ordered(v));
  }
}

// ---------------------------------------------------------------------------
// K4: ex = exp(e - max[dst]); accumulate denom[dst] (native f32 atomic).
// ---------------------------------------------------------------------------
__global__ __launch_bounds__(256) void gat_edge_exp(
    const int* __restrict__ dst, float* __restrict__ escore,
    const unsigned* __restrict__ emax, float* __restrict__ denom, int E)
{
  int e = blockIdx.x * 256 + threadIdx.x;
  if (e >= E) return;
  int d = dst[e];
#pragma unroll
  for (int h = 0; h < 4; ++h) {
    float m  = ordered_to_f32(emax[d * 4 + h]);
    float ex = __expf(escore[(size_t)e * 4 + h] - m);
    escore[(size_t)e * 4 + h] = ex;
    atomic_add_f32_dev(&denom[d * 4 + h], ex);
  }
}

// ---------------------------------------------------------------------------
// K5: weighted scatter: d_out[dst] += feat[src] * attn. 128 threads per edge,
// coalesced gather + native f32 atomic scatter (feat + accumulator are
// L2-resident: 51MB + 51MB << 192MB).
// ---------------------------------------------------------------------------
__global__ __launch_bounds__(256) void gat_edge_scatter(
    const int* __restrict__ src, const int* __restrict__ dst,
    const float* __restrict__ feat, const float* __restrict__ escore,
    const float* __restrict__ denom, float* __restrict__ outx, int E)
{
  long long t = (long long)blockIdx.x * 256 + threadIdx.x;
  int e = (int)(t >> 7);
  if (e >= E) return;
  int j = (int)(t & 127);
  int h = j >> 5;
  int s = src[e], d = dst[e];
  float attn = escore[(size_t)e * 4 + h] / denom[d * 4 + h];
  float v = feat[(size_t)s * 128 + j] * attn;
  atomic_add_f32_dev(&outx[(size_t)d * 128 + j], v);
}

// ---------------------------------------------------------------------------
// Host launcher
// ---------------------------------------------------------------------------
extern "C" void kernel_launch(void* const* d_in, const int* in_sizes, int n_in,
                              void* d_out, int out_size, void* d_ws, size_t ws_size,
                              hipStream_t stream) {
  const float* node_inputs = (const float*)d_in[0];
  const float* edge_inputs = (const float*)d_in[1];
  const int*   src         = (const int*)d_in[2];
  const int*   dst         = (const int*)d_in[3];
  const float* W_fc        = (const float*)d_in[4];
  const float* attn_src    = (const float*)d_in[5];
  const float* attn_dst    = (const float*)d_in[6];
  const float* W_edge      = (const float*)d_in[7];
  const float* attn_edge   = (const float*)d_in[8];
  const float* W_res       = (const float*)d_in[9];
  const float* bias        = (const float*)d_in[10];

  const int N = in_sizes[0] / 128;
  const int E = in_sizes[2];

  char* ws = (char*)d_ws;
  size_t off = 0;
  auto alloc = [&](size_t bytes) -> char* {
    char* p = ws + off;
    off += (bytes + 255) & ~(size_t)255;
    return p;
  };
  _Float16* Ah    = (_Float16*)alloc((size_t)N * 128 * sizeof(_Float16));
  _Float16* Pfc   = (_Float16*)alloc(16384 * sizeof(_Float16));
  _Float16* Pres  = (_Float16*)alloc(16384 * sizeof(_Float16));
  float*    feat  = (float*)   alloc((size_t)N * 128 * sizeof(float));
  float*    a_src = (float*)   alloc((size_t)N * 4 * sizeof(float));
  float*    a_dst = (float*)   alloc((size_t)N * 4 * sizeof(float));
  float*    wsrc  = (float*)   alloc(512 * sizeof(float));
  float*    wdst  = (float*)   alloc(512 * sizeof(float));
  float*    wedge = (float*)   alloc(64 * sizeof(float));
  float*    esc   = (float*)   alloc((size_t)E * 4 * sizeof(float));
  unsigned* emax  = (unsigned*)alloc((size_t)N * 4 * sizeof(unsigned));
  float*    denom = (float*)   alloc((size_t)N * 4 * sizeof(float));
  (void)ws_size; (void)n_in; (void)out_size;

  float* outx = (float*)d_out;

  gat_reduce_attn_w<<<3, 256, 0, stream>>>(W_fc, attn_src, attn_dst, W_edge,
                                           attn_edge, wsrc, wdst, wedge);
  gat_cvt_f16<<<(N * 128 + 255) / 256, 256, 0, stream>>>(node_inputs, Ah, N * 128);
  gat_pack_w<<<128, 256, 0, stream>>>(W_fc, W_res, Pfc, Pres);
  gat_node_proj_wmma<<<(N + 15) / 16, 256, 0, stream>>>(Ah, Pfc, Pres, bias,
                                                        feat, outx, N);
  gat_node_logits<<<(N + 7) / 8, 256, 0, stream>>>(node_inputs, wsrc, wdst,
                                                   a_src, a_dst, emax, denom, N);
  gat_edge_logits<<<(E + 255) / 256, 256, 0, stream>>>(edge_inputs, src, dst,
                                                       wedge, a_src, a_dst,
                                                       esc, emax, E);
  gat_edge_exp<<<(E + 255) / 256, 256, 0, stream>>>(dst, esc, emax, denom, E);
  {
    long long total = (long long)E * 128;
    int blocks = (int)((total + 255) / 256);
    gat_edge_scatter<<<blocks, 256, 0, stream>>>(src, dst, feat, esc, denom,
                                                 outx, E);
  }
}
